// Capsule2Capsule_63814624084387
// MI455X (gfx1250) — compile-verified, matrix operand order checked by reading, not run on hardware
//
#include <hip/hip_runtime.h>
#include <hip/hip_bf16.h>
#include <math.h>

// ---------------------------------------------------------------------------
// CapsNet dynamic routing, fully fused for MI455X (gfx1250, wave32, WMMA).
//
// Shapes: B=64, IC=2048, IH=16, OC=64, OH=32.  U[b,i,o,j] (1 GiB fp32) is
// NEVER materialized: each routing pass recomputes U^T tiles with
// v_wmma_f32_16x16x32_bf16 (A = W-frag, B = x-frag; K padded 16->32 with
// zeros) from a one-time bf16, h-innermost copy of W staged in LDS with
// global_load_async_to_lds_b128 (ASYNCcnt path, no VGPR round trip).
// U^T orientation (lane = batch row, VGPR = (o,j)) makes the U.V reduction
// for the routing logits a pure in-lane sum over 8 VGPRs -- no cross-lane
// shuffles -- and makes V loads / S stores contiguous 32-B per lane.
// Dominant HBM traffic ~= 3 passes * 4 b-tiles * 134 MB ~= 1.6 GB ~= 70 us
// at 23.3 TB/s, vs ~315 us for the naive U-materializing pipeline.
// S is reduced deterministically via per-i-chunk partials (no global atomics).
// ---------------------------------------------------------------------------

typedef __attribute__((ext_vector_type(16))) __bf16 v16bf;
typedef __attribute__((ext_vector_type(8)))  float  v8f;

#define B_    64
#define IC_   2048
#define IH_   16
#define OC_   64
#define OH_   32
#define NOJ   (OC_*OH_)      // 2048 flattened (o,j)
#define BT    16             // batch rows per block  (one WMMA N tile)
#define ICH   16             // input capsules per block
#define NICH  (IC_/ICH)      // 128 i-chunks
#define NWAVE 16             // 512 threads
#define NTPW  8              // (o,j)-tiles (of 16) per wave: 128 / 16 waves
#define WSLAB (NOJ*IH_)      // 32768 bf16 = 64 KB per capsule

// WMMA 16x16x32 bf16 fragment held as 8 contiguous VGPRs: lower uint4 is the
// 8 live bf16 (K = 8*(lane>=16) + e), upper uint4 is the zero K-padding,
// written ONCE so per-tile updates are a single 16-B LDS load into q[0].
union Frag {
  v16bf f;
  uint4 q[2];
};

// X[b,i,h] fp32  ->  Xc[i,b,h] bf16  (fragment friendly: h innermost)
__global__ __launch_bounds__(256)
void cvt_x(const float* __restrict__ X, __bf16* __restrict__ Xc) {
  int q = blockIdx.x * 256 + threadIdx.x;         // < IC_*B_*IH_ = 2M
  int i = q >> 10;                                // / (64*16)
  int b = (q >> 4) & 63;
  int h = q & 15;
  Xc[q] = (__bf16)X[((size_t)b * IC_ + i) * IH_ + h];
}

// W[i,h,oj] fp32  ->  Wc[i,oj,h] bf16  (fragment friendly: h innermost)
__global__ __launch_bounds__(256)
void cvt_w(const float* __restrict__ W, __bf16* __restrict__ Wc) {
  int q  = blockIdx.x * 256 + threadIdx.x;        // < IC_*NOJ = 4.19M
  int i  = q >> 11;
  int oj = q & (NOJ - 1);
  union { uint4 u[2]; __bf16 h[16]; } out;
#pragma unroll
  for (int hh = 0; hh < IH_; ++hh)
    out.h[hh] = (__bf16)W[((size_t)i * IH_ + hh) * NOJ + oj];
  uint4* dst = (uint4*)(Wc + (size_t)q * IH_);
  dst[0] = out.u[0];
  dst[1] = out.u[1];
}

// One fused routing pass.
//   mode 0: C = 1/64 uniform (Blog==0), accumulate S partials.
//   mode 1: dBlog = U.V, Blog := dBlog (write), C = softmax, accumulate S.
//   mode 2: dBlog = U.V, Blog += dBlog (read, no write), C = softmax, acc S.
__global__ __launch_bounds__(512)
void caps_pass(const __bf16* __restrict__ Xc, const __bf16* __restrict__ Wc,
               float* __restrict__ Blog, const float* __restrict__ Vin,
               float* __restrict__ Spart, int mode) {
  __shared__ __bf16 w_ls[WSLAB];          // 64 KB: W slab of current capsule
  __shared__ __bf16 x_ls[ICH * BT * IH_]; //  8 KB: X slab of this block
  __shared__ float  delta_ls[BT * OC_];   //  4 KB: dBlog[b][o]
  __shared__ float  c_ls[BT * OC_];       //  4 KB: C[b][o]

  const int t     = threadIdx.x;
  const int w     = t >> 5;
  const int lane  = t & 31;
  const int nlo   = lane & 15;            // = local batch row b
  const int hi    = lane >> 4;
  const int btile = blockIdx.x;           // 0..3
  const int ich   = blockIdx.y;           // 0..127
  const int b0    = btile * BT;

  // Stage X slab: [il][b_local][h], 512 uint4, one per thread.
  {
    int il = t >> 5, r = t & 31;
    const uint4* src =
        (const uint4*)(Xc + ((size_t)(ich * ICH + il) * B_ + b0) * IH_);
    ((uint4*)x_ls)[il * 32 + r] = src[r];
  }

  // Per-lane LDS byte offset for async staging (low 32 bits of the generic
  // address of a __shared__ object are the LDS offset on gfx1250).
  const uint32_t lds_w0 = (uint32_t)(uintptr_t)(&w_ls[0]) + (uint32_t)(t * 16);

  // Per-wave S accumulators: sacc[tile][v] <-> (b = b0+nlo,
  // oj = (w + NWAVE*tile)*16 + 8*hi + v).  Held across the whole capsule loop.
  v8f sacc[NTPW];
#pragma unroll
  for (int k = 0; k < NTPW; ++k)
#pragma unroll
    for (int v = 0; v < 8; ++v) sacc[k][v] = 0.0f;

  Frag xf, wf;                            // zero K-padding written once
  xf.q[1] = make_uint4(0u, 0u, 0u, 0u);
  wf.q[1] = make_uint4(0u, 0u, 0u, 0u);

  for (int il = 0; il < ICH; ++il) {
    const int ig = ich * ICH + il;
    __syncthreads();                      // w_ls/c_ls no longer in use
    if (mode) { delta_ls[t] = 0.0f; delta_ls[t + 512] = 0.0f; }

    // ---- async-stage the 64 KB W slab: 4096 x 16 B, 8 per thread ----
    {
      const uint64_t gbase =
          (uint64_t)(uintptr_t)(Wc + (size_t)ig * WSLAB);
      const uint32_t goff = (uint32_t)(t * 16);
#pragma unroll
      for (int k = 0; k < 8; ++k) {
        asm volatile("global_load_async_to_lds_b128 %0, %1, %2"
                     :: "v"(lds_w0 + (uint32_t)(k * 8192)),
                        "v"(goff + (uint32_t)(k * 8192)),
                        "s"(gbase)
                     : "memory");
      }
      asm volatile("s_wait_asynccnt 0x0" ::: "memory");
    }
    __syncthreads();

    // B operand: x[b = b0 + nlo, h]; same for every (o,j) tile.
    xf.q[0] = *(const uint4*)&x_ls[(il * BT + nlo) * IH_ + hi * 8];

    if (mode) {
      // ---- phase 1: dBlog[b][o] = sum_j U[b,o,j] * V[b,o,j] ----
      // U^T tile: lane b = nlo, row v -> oj = nt*16 + 8*hi + v.
#pragma unroll
      for (int tt = 0; tt < NTPW; ++tt) {
        const int nt = w + NWAVE * tt;    // 0..127, tile covers one o
        const int o  = nt >> 1;
        wf.q[0] = *(const uint4*)&w_ls[(nt * 16 + nlo) * IH_ + hi * 8];
        v8f u;
#pragma unroll
        for (int v = 0; v < 8; ++v) u[v] = 0.0f;
        u = __builtin_amdgcn_wmma_f32_16x16x32_bf16(
                false, wf.f, false, xf.f, (short)0, u, false, false);
        const float* vp =
            Vin + (size_t)(b0 + nlo) * NOJ + nt * 16 + hi * 8;
        const float4 va = *(const float4*)vp;
        const float4 vb = *(const float4*)(vp + 4);
        float p = u[0] * va.x + u[1] * va.y + u[2] * va.z + u[3] * va.w +
                  u[4] * vb.x + u[5] * vb.y + u[6] * vb.z + u[7] * vb.w;
        unsafeAtomicAdd(&delta_ls[nlo * OC_ + o], p);  // 2-way conflict max
      }
      __syncthreads();

      // ---- Blog update + stage logits ----
      {
        const size_t bbase = ((size_t)ig * B_ + b0) * OC_;
#pragma unroll
        for (int k = 0; k < 2; ++k) {
          int bo = t + 512 * k;
          float val = delta_ls[bo];
          if (mode == 2) val += Blog[bbase + bo];
          else           Blog[bbase + bo] = val;
          c_ls[bo] = val;
        }
      }
      __syncthreads();

      // ---- softmax over o (64) per b row ----
      if (t < BT) {
        float m = -1e30f;
        for (int o = 0; o < OC_; ++o) m = fmaxf(m, c_ls[t * OC_ + o]);
        float s = 0.0f;
        for (int o = 0; o < OC_; ++o) {
          float e = __expf(c_ls[t * OC_ + o] - m);
          c_ls[t * OC_ + o] = e;
          s += e;
        }
        float inv = 1.0f / s;
        for (int o = 0; o < OC_; ++o) c_ls[t * OC_ + o] *= inv;
      }
      __syncthreads();
    }

    // ---- phase 2: sacc += C[b,o] * U^T tile (U recomputed) ----
#pragma unroll
    for (int tt = 0; tt < NTPW; ++tt) {
      const int nt = w + NWAVE * tt;
      const int o  = nt >> 1;
      wf.q[0] = *(const uint4*)&w_ls[(nt * 16 + nlo) * IH_ + hi * 8];
      v8f u;
#pragma unroll
      for (int v = 0; v < 8; ++v) u[v] = 0.0f;
      u = __builtin_amdgcn_wmma_f32_16x16x32_bf16(
              false, wf.f, false, xf.f, (short)0, u, false, false);
      const float c = mode ? c_ls[nlo * OC_ + o] : 0.015625f;
#pragma unroll
      for (int v = 0; v < 8; ++v) sacc[tt][v] += c * u[v];
    }
  }

  // ---- flush S partials: contiguous 32-B stores, no atomics ----
  float* sp = Spart + ((size_t)ich * B_ + b0 + nlo) * NOJ + hi * 8;
#pragma unroll
  for (int tt = 0; tt < NTPW; ++tt) {
    const int nt = w + NWAVE * tt;
    *(v8f*)(sp + nt * 16) = sacc[tt];
  }
}

// Deterministic 128-way reduction of S partials (fully coalesced).
__global__ __launch_bounds__(256)
void reduce_s(const float* __restrict__ Spart, float* __restrict__ S) {
  int t = blockIdx.x * 256 + threadIdx.x;         // < B_*NOJ = 131072
  float acc = 0.0f;
#pragma unroll 8
  for (int ic = 0; ic < NICH; ++ic)
    acc += Spart[(size_t)ic * (B_ * NOJ) + t];
  S[t] = acc;
}

// V[b,o,:] = squash(S[b,o,:])
__global__ __launch_bounds__(256)
void squash_k(const float* __restrict__ S, float* __restrict__ Vout) {
  int bo = blockIdx.x * 256 + threadIdx.x;
  if (bo >= B_ * OC_) return;
  float s[OH_];
  float nsq = 0.0f;
#pragma unroll
  for (int h = 0; h < OH_; ++h) {
    s[h] = S[(size_t)bo * OH_ + h];
    nsq += s[h] * s[h];
  }
  float scale = nsq / ((1.0f + nsq) * sqrtf(nsq));
#pragma unroll
  for (int h = 0; h < OH_; ++h)
    Vout[(size_t)bo * OH_ + h] = s[h] * scale;
}

extern "C" void kernel_launch(void* const* d_in, const int* in_sizes, int n_in,
                              void* d_out, int out_size, void* d_ws,
                              size_t ws_size, hipStream_t stream) {
  const float* X = (const float*)d_in[0];  // [64,2048,16]
  const float* W = (const float*)d_in[1];  // [2048,16,64,32]
  char* ws = (char*)d_ws;
  // ws layout (~229 MB total)
  __bf16* Xc    = (__bf16*)(ws);                 //   4 MB
  __bf16* Wc    = (__bf16*)(ws + 4194304);       // 134 MB
  float*  Blog  = (float*) (ws + 138412032);     //  33.5 MB
  float*  Spart = (float*) (ws + 171966464);     //  64 MB  [128][64][2048]
  float*  S     = (float*) (ws + 239075328);     //   0.5 MB
  float*  V1    = (float*) (ws + 239599616);     //   0.5 MB
  float*  Out   = (float*)d_out;                 // [64,64,32]

  cvt_x<<<(IC_ * B_ * IH_) / 256, 256, 0, stream>>>(X, Xc);
  cvt_w<<<(IC_ * NOJ) / 256, 256, 0, stream>>>(W, Wc);

  dim3 grid(B_ / BT, NICH);                       // (4, 128)
  const int rblk = (B_ * NOJ) / 256;              // 512
  const int qblk = (B_ * OC_) / 256;              // 16

  caps_pass<<<grid, 512, 0, stream>>>(Xc, Wc, Blog, nullptr, Spart, 0);
  reduce_s<<<rblk, 256, 0, stream>>>(Spart, S);
  squash_k<<<qblk, 256, 0, stream>>>(S, V1);

  caps_pass<<<grid, 512, 0, stream>>>(Xc, Wc, Blog, V1, Spart, 1);
  reduce_s<<<rblk, 256, 0, stream>>>(Spart, S);
  squash_k<<<qblk, 256, 0, stream>>>(S, V1);

  caps_pass<<<grid, 512, 0, stream>>>(Xc, Wc, Blog, V1, Spart, 2);
  reduce_s<<<rblk, 256, 0, stream>>>(Spart, S);
  squash_k<<<qblk, 256, 0, stream>>>(S, Out);
}
